// FVSBN_65919158059493
// MI455X (gfx1250) — matrix operand order repaired and checked by reading, not compiled, not run on hardware
//
#include <hip/hip_runtime.h>
#include <hip/hip_bf16.h>

// ---------------------------------------------------------------------------
// out[B,764] = x[B,764] @ (W .* tril(-1))^T + b
// bf16x3 split precision (hi*hi + hi*lo + lo*hi), fp32 WMMA accumulation.
//  - prep kernel: masked W -> packed bf16 hi/lo planes [768][768] in d_ws
//  - gemm kernel: x strip (64 rows) split once into 192KB LDS (hi/lo),
//    then barrier-free N(6 x 128) x K(24 x 32) loops of v_wmma_f32_16x16x32_bf16
//    with B fragments streamed from L2-resident packed W.
// ---------------------------------------------------------------------------

typedef __attribute__((ext_vector_type(16))) __bf16 v16bf;
typedef __attribute__((ext_vector_type(8)))  __bf16 v8bf;
typedef __attribute__((ext_vector_type(8)))  float  v8f;

#define D_DIM 764
#define DP    768          // padded K / N extent
#define KT    32           // K per WMMA chunk
#define NKT   24           // 768 / 32
#define BM    64           // rows per block
#define BN    128          // cols per N-tile iteration
#define NNT   6            // 768 / 128

__device__ __forceinline__ unsigned short f2bf(float f) {
  // round-to-nearest-even fp32 -> bf16 (finite inputs)
  unsigned u = __builtin_bit_cast(unsigned, f);
  unsigned r = u + 0x7FFFu + ((u >> 16) & 1u);
  return (unsigned short)(r >> 16);
}
__device__ __forceinline__ float bf2f(unsigned short h) {
  return __builtin_bit_cast(float, (unsigned)h << 16);
}

// ---------------- prep: masked W -> bf16 hi/lo planes in workspace ----------
__global__ __launch_bounds__(256)
void fvsbn_prep_w(const float* __restrict__ W,
                  unsigned short* __restrict__ Whi,
                  unsigned short* __restrict__ Wlo)
{
  int idx = blockIdx.x * 256 + threadIdx.x;   // over DP*DP
  int n = idx / DP;
  int k = idx % DP;
  float v = 0.f;
  if (n < D_DIM && k < n)                      // strict lower triangle
    v = W[(size_t)n * D_DIM + k];              // k < n <= 763 -> in range
  unsigned short h = f2bf(v);
  unsigned short l = f2bf(v - bf2f(h));
  Whi[idx] = h;
  Wlo[idx] = l;
}

// ---------------- main GEMM ------------------------------------------------
__global__ __launch_bounds__(256)
void fvsbn_gemm(const float* __restrict__ x,
                const unsigned short* __restrict__ Whi,   // [DP][DP]
                const unsigned short* __restrict__ Wlo,   // [DP][DP]
                const float* __restrict__ bias,
                float* __restrict__ out)
{
  __shared__ __align__(32) unsigned short Ahi[BM][DP];    // 96 KB
  __shared__ __align__(32) unsigned short Alo[BM][DP];    // 96 KB

  const int t     = threadIdx.x;
  const int mBase = blockIdx.x * BM;
  const int lane  = t & 31;
  const int wave  = t >> 5;
  const int mOff  = (wave >> 2) * 32;    // 2 waves along M
  const int nOff  = (wave & 3)  * 32;    // 4 waves along N (within BN)
  const int fr    = lane & 15;           // row within 16x16 fragment
  const int grp   = lane >> 4;           // lane half-group

  // ---- stage x strip -> LDS bf16 hi/lo (once per block) ----
  #pragma unroll 4
  for (int i = 0; i < (BM * DP) / (256 * 4); ++i) {   // 48 float4 per thread
    int ff  = t + i * 256;            // flat float4 index
    int row = ff / (DP / 4);          // DP/4 = 192 float4 per row
    int c4  = ff % (DP / 4);
    int col = c4 * 4;
    float4 v = make_float4(0.f, 0.f, 0.f, 0.f);
    if (col < D_DIM)                  // col%4==0 and 764%4==0 -> whole float4 valid
      v = *(const float4*)(x + (size_t)(mBase + row) * D_DIM + col);
    ushort4 h, l;
    h.x = f2bf(v.x); l.x = f2bf(v.x - bf2f(h.x));
    h.y = f2bf(v.y); l.y = f2bf(v.y - bf2f(h.y));
    h.z = f2bf(v.z); l.z = f2bf(v.z - bf2f(h.z));
    h.w = f2bf(v.w); l.w = f2bf(v.w - bf2f(h.w));
    *(ushort4*)&Ahi[row][col] = h;
    *(ushort4*)&Alo[row][col] = l;
  }
  __syncthreads();   // LDS read-only from here: no further barriers needed

  // ---- N-tile loop: 6 x 128 output columns ----
  #pragma unroll 1
  for (int nt = 0; nt < NNT; ++nt) {
    const int nTile = nt * BN;
    v8f acc[2][2] = {};

    // B fragments stream from global (L2-resident), double-buffered on K
    v16bf bh[2][2], bl[2][2];   // [parity][tn]
    auto loadB = [&](int kt, int par) {
      #pragma unroll
      for (int tn = 0; tn < 2; ++tn) {
        int n = nTile + nOff + tn * 16 + fr;            // < 768
        size_t off = (size_t)n * DP + kt * KT + grp * 16;
        bh[par][tn] = *(const v16bf*)(Whi + off);
        bl[par][tn] = *(const v16bf*)(Wlo + off);
      }
    };
    loadB(0, 0);

    #pragma unroll 2
    for (int kt = 0; kt < NKT; ++kt) {
      const int par = kt & 1;
      if (kt + 1 < NKT) loadB(kt + 1, par ^ 1);         // prefetch next chunk

      // A fragments (16-bit 16x32): half-group g holds
      // K = {16g..16g+15} relative to the chunk base (matches B).
      const int kBase = kt * KT;
      v16bf ah[2], al[2];
      #pragma unroll
      for (int tm = 0; tm < 2; ++tm) {
        int r = mOff + tm * 16 + fr;
        v8bf h0 = *(const v8bf*)&Ahi[r][kBase + grp * 16];
        v8bf h1 = *(const v8bf*)&Ahi[r][kBase + grp * 16 + 8];
        ah[tm] = __builtin_shufflevector(h0, h1, 0,1,2,3,4,5,6,7,8,9,10,11,12,13,14,15);
        v8bf l0 = *(const v8bf*)&Alo[r][kBase + grp * 16];
        v8bf l1 = *(const v8bf*)&Alo[r][kBase + grp * 16 + 8];
        al[tm] = __builtin_shufflevector(l0, l1, 0,1,2,3,4,5,6,7,8,9,10,11,12,13,14,15);
      }

      // bf16x3: hi*hi + hi*lo + lo*hi
      #pragma unroll
      for (int tm = 0; tm < 2; ++tm)
        #pragma unroll
        for (int tn = 0; tn < 2; ++tn) {
          acc[tm][tn] = __builtin_amdgcn_wmma_f32_16x16x32_bf16(
              false, ah[tm], false, bh[par][tn], (short)0, acc[tm][tn], false, false);
          acc[tm][tn] = __builtin_amdgcn_wmma_f32_16x16x32_bf16(
              false, ah[tm], false, bl[par][tn], (short)0, acc[tm][tn], false, false);
          acc[tm][tn] = __builtin_amdgcn_wmma_f32_16x16x32_bf16(
              false, al[tm], false, bh[par][tn], (short)0, acc[tm][tn], false, false);
        }
    }

    // ---- epilogue for this N-tile: C layout N=lane%16, M=r+8*(lane/16) ----
    #pragma unroll
    for (int tn = 0; tn < 2; ++tn) {
      int n = nTile + nOff + tn * 16 + fr;
      if (n >= D_DIM) continue;
      float bv = bias[n];
      #pragma unroll
      for (int tm = 0; tm < 2; ++tm) {
        int mrow = mBase + mOff + tm * 16 + (grp << 3);
        #pragma unroll
        for (int r = 0; r < 8; ++r) {
          out[(size_t)(mrow + r) * D_DIM + n] = acc[tm][tn][r] + bv;
        }
      }
    }
  }
}

extern "C" void kernel_launch(void* const* d_in, const int* in_sizes, int n_in,
                              void* d_out, int out_size, void* d_ws, size_t ws_size,
                              hipStream_t stream) {
  (void)n_in; (void)out_size; (void)ws_size;
  const float* x = (const float*)d_in[0];
  const float* W = (const float*)d_in[1];
  const float* b = (const float*)d_in[2];
  float* out     = (float*)d_out;

  unsigned short* Whi = (unsigned short*)d_ws;           // DP*DP ushort
  unsigned short* Wlo = Whi + (size_t)DP * DP;           // DP*DP ushort (total 2.36 MB)

  // prep masked bf16 hi/lo W planes
  fvsbn_prep_w<<<(DP * DP) / 256, 256, 0, stream>>>(W, Whi, Wlo);

  // main GEMM: one block per 64-row strip, full N inside
  const int Brows = in_sizes[0] / D_DIM;                 // 65536
  fvsbn_gemm<<<Brows / BM, 256, 0, stream>>>(x, Whi, Wlo, b, out);
}
